// XOnlyGNN_45732811768303
// MI455X (gfx1250) — compile-verified, matrix operand order checked by reading, not run on hardware
//
#include <hip/hip_runtime.h>
#include <stdint.h>

#define H       64
#define LAYERS  2
#define N_AP    512
#define N_USER  4096
#define NEDGE   32768
#define D_AP    7
#define D_USER  2
#define D_EDGE  6

typedef __attribute__((ext_vector_type(16))) __bf16       v16bf;
typedef __attribute__((ext_vector_type(8)))  float        v8f;
typedef __attribute__((ext_vector_type(4)))  unsigned int u32x4;
typedef __attribute__((ext_vector_type(8)))  unsigned int u32x8;
typedef int v4i_vs __attribute__((vector_size(16)));   // matches builtin param type

__device__ __forceinline__ unsigned short f2bf(float f) {
    unsigned int u = __float_as_uint(f);
    u += 0x7fffu + ((u >> 16) & 1u);   // round-to-nearest-even
    return (unsigned short)(u >> 16);
}

// ---- CDNA5 async global->LDS copy (ASYNCcnt-tracked), 16B per lane ----------
__device__ __forceinline__ void async_copy16(uint32_t lds_off, const uint32_t* gp) {
#if __has_builtin(__builtin_amdgcn_global_load_async_to_lds_b128)
    __builtin_amdgcn_global_load_async_to_lds_b128(
        (__attribute__((address_space(1))) v4i_vs*)(uintptr_t)gp,
        (__attribute__((address_space(3))) v4i_vs*)lds_off,
        0, 0);
#else
    asm volatile("global_load_async_to_lds_b128 %0, %1, off"
                 :: "v"(lds_off), "v"(gp) : "memory");
#endif
}

template <int N>
__device__ __forceinline__ void wait_async() {
#if __has_builtin(__builtin_amdgcn_s_wait_asynccnt)
    __builtin_amdgcn_s_wait_asynccnt((unsigned short)N);
#else
    asm volatile("s_wait_asynccnt %0" :: "n"(N));
#endif
}

// ---------------------------------------------------------------- utilities
__global__ void zero_f32(float* __restrict__ p, int n) {
    int i = blockIdx.x * blockDim.x + threadIdx.x;
    if (i < n) p[i] = 0.f;
}

// Pre-pack We2 (fp32 [L,64,4096]) into bf16 WMMA B-operand register layout.
// dest dword index = ((((l*64+h)*4+ot)*2+kk)*32 + lane)*8 + r
__global__ void pack_we2(const float* __restrict__ We2, uint32_t* __restrict__ bp) {
    int idx = blockIdx.x * blockDim.x + threadIdx.x;
    int r    = idx & 7;        int t = idx >> 3;
    int lane = t & 31;         t >>= 5;
    int kk   = t & 1;          t >>= 1;
    int ot   = t & 3;          t >>= 2;
    int h    = t & 63;         t >>= 6;
    int l    = t;
    if (l >= LAYERS) return;
    int hi = lane >> 4, n = lane & 15;
    int kg = (r < 4) ? (2 * r) : (16 + 2 * (r - 4));
    int k0 = kg + hi * 8 + kk * 32;          // even K of the dword pair
    int c  = h * 64 + ot * 16 + n;           // column = h*H + o
    const float* W = We2 + (size_t)l * 64 * 4096;
    unsigned int w0 = f2bf(W[(size_t)k0       * 4096 + c]);
    unsigned int w1 = f2bf(W[(size_t)(k0 + 1) * 4096 + c]);
    bp[idx] = w0 | (w1 << 16);
}

// ------------------------------------------------------------ small MLPs
__global__ void encode_mlp(const float* __restrict__ x, int D,
                           const float* __restrict__ W1, const float* __restrict__ b1,
                           const float* __restrict__ W2, const float* __restrict__ b2,
                           float* __restrict__ out) {
    __shared__ float xr[8];
    __shared__ float hid[H];
    int n = blockIdx.x, t = threadIdx.x;
    if (t < D) xr[t] = x[(size_t)n * D + t];
    __syncthreads();
    float a = b1[t];
    for (int d = 0; d < D; ++d) a += xr[d] * W1[d * H + t];
    hid[t] = fmaxf(a, 0.f);
    __syncthreads();
    float o = b2[t];
#pragma unroll
    for (int hh = 0; hh < H; ++hh) o += hid[hh] * W2[hh * H + t];
    out[(size_t)n * H + t] = o;
}

__global__ void count_deg(const int* __restrict__ src, const int* __restrict__ dst,
                          float* __restrict__ cntA, float* __restrict__ cntU) {
    int e = blockIdx.x * blockDim.x + threadIdx.x;
    if (e < NEDGE) {
        atomicAdd(cntA + src[e], 1.f);
        atomicAdd(cntU + dst[e], 1.f);
    }
}

// ------------------------------------------------- fused edge-conditioned conv
// Per wave: 16 edges. Z[16x64]=relu(EA@We1+be1) as bf16 A-operand. The packed
// We2 h-slice (8KB) is async-copied global->LDS ONCE per block (double
// buffered, overlapped with WMMA), then each wave runs 8 v_wmma per h and
// accumulates msg_u/msg_a with per-row scales ha_src[e,h] / hu_dst[e,h].
#define MSG_WAVES 4
__global__ __launch_bounds__(MSG_WAVES * 32)
void edge_msg(const float* __restrict__ ha, const float* __restrict__ hu,
              const float* __restrict__ edge_attr,
              const int* __restrict__ src, const int* __restrict__ dst,
              const float* __restrict__ We1, const float* __restrict__ be1,
              const uint32_t* __restrict__ bpk, const float* __restrict__ be2,
              float* __restrict__ aggU, float* __restrict__ aggA) {
    __shared__ float                        haS[MSG_WAVES][16][H];
    __shared__ float                        huD[MSG_WAVES][16][H];
    __shared__ __align__(16) unsigned short zl [MSG_WAVES][16][H];
    __shared__ __align__(16) uint32_t       bst[2][2048];   // 2 x 8KB B staging

    int tid  = threadIdx.x;
    int wave = tid >> 5, lane = tid & 31;
    int e0 = (blockIdx.x * MSG_WAVES + wave) * 16;

    // ---- kick off async copy of B tiles for h=0 (16 dwords per thread)
    {
        uint32_t l0 = (uint32_t)(uintptr_t)&bst[0][tid * 16];
        const uint32_t* g0 = bpk + tid * 16;
#pragma unroll
        for (int j = 0; j < 4; ++j) async_copy16(l0 + j * 16, g0 + j * 4);
    }

    // gather ha[src], hu[dst] rows for the 16 edges
    for (int i = lane; i < 16 * H; i += 32) {
        int e = i >> 6, hh = i & 63;
        haS[wave][e][hh] = ha[(size_t)src[e0 + e] * H + hh];
        huD[wave][e][hh] = hu[(size_t)dst[e0 + e] * H + hh];
    }
    // Z = relu(edge_attr @ We1 + be1) -> bf16
    for (int i = lane; i < 16 * H; i += 32) {
        int e = i >> 6, k = i & 63;
        float a = be1[k];
#pragma unroll
        for (int d = 0; d < D_EDGE; ++d)
            a += edge_attr[(size_t)(e0 + e) * D_EDGE + d] * We1[d * H + k];
        zl[wave][e][k] = f2bf(fmaxf(a, 0.f));
    }
    __syncthreads();

    // pack Z into A-operand layout: lanes 0-15 hold K {0..7,16..23}(+kk*32),
    // lanes 16-31 the +8 variants -> contiguous 16B runs of the row-major row.
    int M = lane & 15, hi = lane >> 4, n = M;
    const unsigned short* zr = zl[wave][M];
    u32x4 p0 = *(const u32x4*)(zr + hi * 8);
    u32x4 p1 = *(const u32x4*)(zr + 16 + hi * 8);
    u32x4 p2 = *(const u32x4*)(zr + 32 + hi * 8);
    u32x4 p3 = *(const u32x4*)(zr + 48 + hi * 8);
    v16bf a0 = __builtin_bit_cast(v16bf, __builtin_shufflevector(p0, p1, 0,1,2,3,4,5,6,7));
    v16bf a1 = __builtin_bit_cast(v16bf, __builtin_shufflevector(p2, p3, 0,1,2,3,4,5,6,7));

    float msgU[4][8], msgA[4][8];
#pragma unroll
    for (int ot = 0; ot < 4; ++ot)
#pragma unroll
        for (int r = 0; r < 8; ++r) { msgU[ot][r] = 0.f; msgA[ot][r] = 0.f; }

    for (int h = 0; h < H; ++h) {
        int p = h & 1;
        if (h + 1 < H) {   // prefetch next h-slice into the other buffer
            uint32_t l0 = (uint32_t)(uintptr_t)&bst[p ^ 1][tid * 16];
            const uint32_t* g0 = bpk + (size_t)(h + 1) * 2048 + tid * 16;
#pragma unroll
            for (int j = 0; j < 4; ++j) async_copy16(l0 + j * 16, g0 + j * 4);
            wait_async<4>();   // in-order: the 4 copies for h are complete
        } else {
            wait_async<0>();
        }
        __syncthreads();       // buffer p fully populated by all waves

        float su[8], sa[8];
#pragma unroll
        for (int r = 0; r < 8; ++r) {           // C row M = r + 8*hi
            su[r] = haS[wave][r + 8 * hi][h];
            sa[r] = huD[wave][r + 8 * hi][h];
        }
#pragma unroll
        for (int ot = 0; ot < 4; ++ot) {
            u32x8 b0u = *(const u32x8*)&bst[p][((ot * 2 + 0) * 32 + lane) * 8];
            u32x8 b1u = *(const u32x8*)&bst[p][((ot * 2 + 1) * 32 + lane) * 8];
            v16bf b0 = __builtin_bit_cast(v16bf, b0u);
            v16bf b1 = __builtin_bit_cast(v16bf, b1u);
            v8f acc = {};
            acc = __builtin_amdgcn_wmma_f32_16x16x32_bf16(false, a0, false, b0,
                                                          (short)0, acc, false, false);
            acc = __builtin_amdgcn_wmma_f32_16x16x32_bf16(false, a1, false, b1,
                                                          (short)0, acc, false, false);
            float bias = be2[h * 64 + ot * 16 + n];
#pragma unroll
            for (int r = 0; r < 8; ++r) {
                float v = acc[r] + bias;
                msgU[ot][r] += su[r] * v;
                msgA[ot][r] += sa[r] * v;
            }
        }
        __syncthreads();       // release buffer p before it is overwritten
    }

    // scatter-add messages
#pragma unroll
    for (int r = 0; r < 8; ++r) {
        int e  = e0 + r + 8 * hi;
        int du = dst[e], da = src[e];
#pragma unroll
        for (int ot = 0; ot < 4; ++ot) {
            atomicAdd(aggU + (size_t)du * H + ot * 16 + n, msgU[ot][r]);
            atomicAdd(aggA + (size_t)da * H + ot * 16 + n, msgA[ot][r]);
        }
    }
}

// h_out = relu(agg/cnt + h_in @ root + bias)
__global__ void node_update(const float* __restrict__ h_in, const float* __restrict__ agg,
                            const float* __restrict__ cnt, const float* __restrict__ root,
                            const float* __restrict__ bias, float* __restrict__ h_out) {
    __shared__ float row[H];
    int nID = blockIdx.x, t = threadIdx.x;
    row[t] = h_in[(size_t)nID * H + t];
    __syncthreads();
    float a = bias[t] + agg[(size_t)nID * H + t] / fmaxf(cnt[nID], 1.f);
#pragma unroll
    for (int hh = 0; hh < H; ++hh) a += row[hh] * root[hh * H + t];
    h_out[(size_t)nID * H + t] = fmaxf(a, 0.f);
}

// x_logit = relu([ha_src | hu_dst | ea] @ Wh1 + bh1) @ Wh2 + bh2
__global__ void edge_head(const float* __restrict__ ha, const float* __restrict__ hu,
                          const float* __restrict__ edge_attr,
                          const int* __restrict__ src, const int* __restrict__ dst,
                          const float* __restrict__ Wh1, const float* __restrict__ bh1,
                          const float* __restrict__ Wh2, const float* __restrict__ bh2,
                          float* __restrict__ out) {
    __shared__ float fa[H], fu[H], ea[8], hid[H];
    int e = blockIdx.x, t = threadIdx.x;
    fa[t] = ha[(size_t)src[e] * H + t];
    fu[t] = hu[(size_t)dst[e] * H + t];
    if (t < D_EDGE) ea[t] = edge_attr[(size_t)e * D_EDGE + t];
    __syncthreads();
    float a = bh1[t];
#pragma unroll
    for (int i = 0; i < H; ++i) a += fa[i] * Wh1[i * H + t];
#pragma unroll
    for (int i = 0; i < H; ++i) a += fu[i] * Wh1[(H + i) * H + t];
#pragma unroll
    for (int d = 0; d < D_EDGE; ++d) a += ea[d] * Wh1[(2 * H + d) * H + t];
    hid[t] = fmaxf(a, 0.f);
    __syncthreads();
    if (t == 0) {
        float s = bh2[0];
        for (int i = 0; i < H; ++i) s += hid[i] * Wh2[i];
        out[e] = s;
    }
}

__global__ void node_head(const float* __restrict__ hn,
                          const float* __restrict__ W1, const float* __restrict__ b1,
                          const float* __restrict__ W2, const float* __restrict__ b2,
                          float* __restrict__ out) {
    __shared__ float row[H], hid[H];
    int nID = blockIdx.x, t = threadIdx.x;
    row[t] = hn[(size_t)nID * H + t];
    __syncthreads();
    float a = b1[t];
#pragma unroll
    for (int i = 0; i < H; ++i) a += row[i] * W1[i * H + t];
    hid[t] = fmaxf(a, 0.f);
    __syncthreads();
    if (t == 0) {
        float s = b2[0];
        for (int i = 0; i < H; ++i) s += hid[i] * W2[i];
        out[nID] = s;
    }
}

// ---------------------------------------------------------------- launcher
extern "C" void kernel_launch(void* const* d_in, const int* in_sizes, int n_in,
                              void* d_out, int out_size, void* d_ws, size_t ws_size,
                              hipStream_t stream) {
    (void)in_sizes; (void)n_in; (void)out_size; (void)ws_size;
    const float* x_ap      = (const float*)d_in[0];
    const float* x_user    = (const float*)d_in[1];
    const float* edge_attr = (const float*)d_in[2];
    const int*   ei_src    = (const int*)d_in[3];
    const int*   ei_dst    = (const int*)d_in[4];
    const float* Wa1 = (const float*)d_in[5];  const float* ba1 = (const float*)d_in[6];
    const float* Wa2 = (const float*)d_in[7];  const float* ba2 = (const float*)d_in[8];
    const float* Wu1 = (const float*)d_in[9];  const float* bu1 = (const float*)d_in[10];
    const float* Wu2 = (const float*)d_in[11]; const float* bu2 = (const float*)d_in[12];
    const float* We1 = (const float*)d_in[13]; const float* be1 = (const float*)d_in[14];
    const float* We2 = (const float*)d_in[15]; const float* be2 = (const float*)d_in[16];
    const float* root_f = (const float*)d_in[17]; const float* bias_f = (const float*)d_in[18];
    const float* root_r = (const float*)d_in[19]; const float* bias_r = (const float*)d_in[20];
    const float* Wh1 = (const float*)d_in[21]; const float* bh1 = (const float*)d_in[22];
    const float* Wh2 = (const float*)d_in[23]; const float* bh2 = (const float*)d_in[24];
    const float* Wp1 = (const float*)d_in[25]; const float* bp1 = (const float*)d_in[26];
    const float* Wp2 = (const float*)d_in[27]; const float* bp2 = (const float*)d_in[28];
    float* out = (float*)d_out;

    // workspace carve-up (floats); bpk first for 32B alignment of B chunks
    const size_t BPK_DW = (size_t)LAYERS * 64 * 4 * 2 * 32 * 8;  // 262144 dwords
    uint32_t* bpk = (uint32_t*)d_ws;
    float* ws  = (float*)d_ws;
    size_t off = BPK_DW;
    float* ha0  = ws + off; off += (size_t)N_AP * H;
    float* ha1  = ws + off; off += (size_t)N_AP * H;
    float* hu0  = ws + off; off += (size_t)N_USER * H;
    float* hu1  = ws + off; off += (size_t)N_USER * H;
    float* cntU = ws + off; off += N_USER;
    float* cntA = ws + off; off += N_AP;          // contiguous with cntU
    float* aggU = ws + off; off += (size_t)N_USER * H;
    float* aggA = ws + off; off += (size_t)N_AP * H;   // contiguous with aggU

    pack_we2<<<(int)(BPK_DW / 256), 256, 0, stream>>>(We2, bpk);
    encode_mlp<<<N_AP,   H, 0, stream>>>(x_ap,   D_AP,   Wa1, ba1, Wa2, ba2, ha0);
    encode_mlp<<<N_USER, H, 0, stream>>>(x_user, D_USER, Wu1, bu1, Wu2, bu2, hu0);
    zero_f32<<<(N_USER + N_AP + 255) / 256, 256, 0, stream>>>(cntU, N_USER + N_AP);
    count_deg<<<NEDGE / 256, 256, 0, stream>>>(ei_src, ei_dst, cntA, cntU);

    const float* haIn = ha0; float* haOut = ha1;
    const float* huIn = hu0; float* huOut = hu1;
    for (int l = 0; l < LAYERS; ++l) {
        zero_f32<<<((N_USER + N_AP) * H + 255) / 256, 256, 0, stream>>>(aggU,
                                                                        (N_USER + N_AP) * H);
        edge_msg<<<NEDGE / 16 / MSG_WAVES, MSG_WAVES * 32, 0, stream>>>(
            haIn, huIn, edge_attr, ei_src, ei_dst,
            We1 + (size_t)l * D_EDGE * H, be1 + (size_t)l * H,
            bpk + (size_t)l * 64 * 4 * 2 * 32 * 8, be2 + (size_t)l * H * H,
            aggU, aggA);
        node_update<<<N_USER, H, 0, stream>>>(huIn, aggU, cntU,
                                              root_f + (size_t)l * H * H,
                                              bias_f + (size_t)l * H, huOut);
        node_update<<<N_AP, H, 0, stream>>>(haIn, aggA, cntA,
                                            root_r + (size_t)l * H * H,
                                            bias_r + (size_t)l * H, haOut);
        const float* t0 = haIn; haIn = haOut; haOut = (float*)t0;
        const float* t1 = huIn; huIn = huOut; huOut = (float*)t1;
    }

    edge_head<<<NEDGE, H, 0, stream>>>(haIn, huIn, edge_attr, ei_src, ei_dst,
                                       Wh1, bh1, Wh2, bh2, out);
    node_head<<<N_AP, H, 0, stream>>>(haIn, Wp1, bp1, Wp2, bp2, out + NEDGE);
}
